// TransformerBlock_81870666597080
// MI455X (gfx1250) — matrix operand order repaired and checked by reading, not compiled, not run on hardware
//
#include <hip/hip_runtime.h>

#define D_MODEL 1024
#define NHEAD   16
#define DHEAD   64
#define SEQ     2048
#define BATCH   2
#define MTOK    (BATCH * SEQ)     /* 4096 rows of tokens */
#define HIDDEN  (4 * D_MODEL)     /* 4096 */
#define QKV_N   (3 * D_MODEL)     /* 3072 */

typedef __attribute__((ext_vector_type(16))) __bf16 v16bf;
typedef __attribute__((ext_vector_type(8)))  float  v8f;
typedef unsigned int u32x4 __attribute__((ext_vector_type(4)));
typedef int          i32x4 __attribute__((ext_vector_type(4)));
typedef int          i32x8 __attribute__((ext_vector_type(8)));

union Frag {
    v16bf v;
    uint4 q[2];
    unsigned short u[16];
};

__device__ __forceinline__ unsigned short f2bf(float f) {
    unsigned int u = __float_as_uint(f);
    u += 0x7fffu + ((u >> 16) & 1u);          // round-to-nearest-even
    return (unsigned short)(u >> 16);
}
__device__ __forceinline__ float bf2f(unsigned short h) {
    return __uint_as_float(((unsigned int)h) << 16);
}

// ---------------------------------------------------------------------------
// CDNA5 async global->LDS copy (GLOBAL_LOAD_ASYNC_TO_LDS_B128, ASYNCcnt).
// Per-lane: 16 bytes from the lane's global address to the lane's LDS address.
// Generic LDS pointers carry the LDS byte offset in their low 32 bits.
// ---------------------------------------------------------------------------
__device__ __forceinline__ void async_load_b128(void* lds_dst, const void* gsrc) {
    asm volatile("global_load_async_to_lds_b128 %0, %1, off"
                 :: "v"((unsigned)(size_t)lds_dst),
                    "v"((unsigned long long)(size_t)gsrc)
                 : "memory");
}
__device__ __forceinline__ void wait_async0() {
#if __has_builtin(__builtin_amdgcn_s_wait_asynccnt)
    __builtin_amdgcn_s_wait_asynccnt(0);
#else
    asm volatile("s_wait_asynccnt 0" ::: "memory");
#endif
}
__device__ __forceinline__ void wait_tensor0() {
#if __has_builtin(__builtin_amdgcn_s_wait_tensorcnt)
    __builtin_amdgcn_s_wait_tensorcnt(0);
#else
    asm volatile("s_wait_tensorcnt 0" ::: "memory");
#endif
}

// ---------------------------------------------------------------------------
// Tensor Data Mover: DMA a 2-D bf16 tile (tile_k x tile_rows) from a row-major
// tensor (row length tensor_k elems, stride tensor_k) straight into LDS, rows
// packed contiguously.  D# built per CDNA5 ISA section 8 (groups 2/3 zero for
// 2-D tensors, workgroup_mask=0 outside clusters).  Issued by one wave;
// completion tracked by TENSORcnt.
// ---------------------------------------------------------------------------
#if __has_builtin(__builtin_amdgcn_tensor_load_to_lds)
#define HAVE_TDM 1
__device__ __forceinline__ void tdm_load_tile_2d(void* lds_dst, const void* gsrc,
                                                 unsigned tile_k, unsigned tile_rows,
                                                 unsigned tensor_k, unsigned tensor_rows) {
    const unsigned long long ga = (unsigned long long)(size_t)gsrc;
    u32x4 g0;
    g0[0] = 1u;                                                  // count=1 (valid)
    g0[1] = (unsigned)(size_t)lds_dst;                           // lds_addr (bytes)
    g0[2] = (unsigned)ga;                                        // global_addr[31:0]
    g0[3] = (unsigned)((ga >> 32) & 0x01ffffffu) | 0x80000000u;  // addr[56:32] | type=2
    i32x8 g1;
    g1[0] = 0x00010000;                                          // data_size=1 (2B), mask=0
    g1[1] = (int)(tensor_k << 16);                               // tensor_dim0 bits[15:0]
    g1[2] = (int)((tensor_k >> 16) | (tensor_rows << 16));       // dim0 hi | dim1 lo
    g1[3] = (int)((tensor_rows >> 16) | (tile_k << 16));         // dim1 hi | tile_dim0
    g1[4] = (int)(tile_rows & 0xffffu);                          // tile_dim1 (tile_dim2=0)
    g1[5] = (int)tensor_k;                                       // dim0_stride[31:0]
    g1[6] = (int)(tensor_k >> 16);                               // dim0_stride[47:32]
    g1[7] = 0;                                                   // dim1_stride hi
    const i32x4 z4 = {0, 0, 0, 0};
#if __clang_major__ >= 23
    const i32x8 z8 = {0, 0, 0, 0, 0, 0, 0, 0};
    __builtin_amdgcn_tensor_load_to_lds(g0, g1, z4, z4, z8, 0);
#else
    __builtin_amdgcn_tensor_load_to_lds(g0, g1, z4, z4, 0);
#endif
}
#else
#define HAVE_TDM 0
#endif

// ---------------------------------------------------------------------------
// WMMA fragment loaders, following CDNA5 ISA 7.12.2 VGPR layouts (wave32).
// A (16x32 bf16, MxK): lane = row(M)=lane&15; lanes 0-15 hold K {0-7,16-23},
// lanes 16-31 hold K {8-15,24-31}.  Source is row-major [row][k], ld elems.
// ---------------------------------------------------------------------------
__device__ __forceinline__ v16bf load_a_frag(const unsigned short* base, int ld, int lane) {
    const int row = lane & 15;
    const int hi  = (lane >> 4) & 1;
    const unsigned short* p = base + (size_t)row * ld + hi * 8;
    Frag f;
    f.q[0] = *(const uint4*)(p);        // K +0..7   (+8 when hi)
    f.q[1] = *(const uint4*)(p + 16);   // K +16..23 (+24 when hi)
    return f.v;
}
// B (32x16 bf16, KxN): lane = col(N)=lane&15; lanes 0-15 hold K {0-7,16-23},
// lanes 16-31 hold K {8-15,24-31}.  Source layout is [n][k] row-major (k contiguous).
__device__ __forceinline__ v16bf load_b_frag(const unsigned short* base, int ld, int lane) {
    const int col = lane & 15;
    const int hi  = (lane >> 4) & 1;
    const unsigned short* p = base + (size_t)col * ld + hi * 8;
    Frag f;
    f.q[0] = *(const uint4*)(p);
    f.q[1] = *(const uint4*)(p + 16);
    return f.v;
}

__device__ __forceinline__ v8f wmma_bf16(v16bf a, v16bf b, v8f c) {
    return __builtin_amdgcn_wmma_f32_16x16x32_bf16(false, a, false, b, (short)0, c,
                                                   false, false);
}

// ---------------------------------------------------------------------------
// fp32 -> bf16 conversion (weights, grid-stride)
// ---------------------------------------------------------------------------
__global__ void cvt_f32_bf16_kernel(const float* __restrict__ in,
                                    unsigned short* __restrict__ out, int n) {
    for (int i = blockIdx.x * blockDim.x + threadIdx.x; i < n;
         i += gridDim.x * blockDim.x)
        out[i] = f2bf(in[i]);
}

// ---------------------------------------------------------------------------
// RMSNorm: fp32 [rows, 1024] -> bf16, one block (256 thr) per row
// ---------------------------------------------------------------------------
__global__ __launch_bounds__(256) void rmsnorm_kernel(const float* __restrict__ x,
                                                      const float* __restrict__ scale,
                                                      unsigned short* __restrict__ out) {
    __shared__ float red[256];
    const int row = blockIdx.x;
    const int tid = threadIdx.x;
    const float4 v = *(const float4*)(x + (size_t)row * D_MODEL + tid * 4);
    red[tid] = v.x * v.x + v.y * v.y + v.z * v.z + v.w * v.w;
    __syncthreads();
    for (int s = 128; s > 0; s >>= 1) {
        if (tid < s) red[tid] += red[tid + s];
        __syncthreads();
    }
    const float r = rsqrtf(red[0] * (1.0f / D_MODEL) + 1e-6f);
    const float4 sc = *(const float4*)(scale + tid * 4);
    ushort4 o;
    o.x = f2bf(v.x * r * sc.x);
    o.y = f2bf(v.y * r * sc.y);
    o.z = f2bf(v.z * r * sc.z);
    o.w = f2bf(v.w * r * sc.w);
    *(ushort4*)(out + (size_t)row * D_MODEL + tid * 4) = o;
}

// ---------------------------------------------------------------------------
// Transpose V head-slices: qkv[b,s, 2048 + h*64 + d] -> vt[b,h,d,s] (bf16)
// makes the P@V B-fragment loads contiguous.
// ---------------------------------------------------------------------------
__global__ void transpose_v_kernel(const unsigned short* __restrict__ qkv,
                                   unsigned short* __restrict__ vt) {
    const size_t total = (size_t)BATCH * NHEAD * DHEAD * SEQ;
    for (size_t i = blockIdx.x * (size_t)blockDim.x + threadIdx.x; i < total;
         i += gridDim.x * (size_t)blockDim.x) {
        const int s_ = (int)(i & (SEQ - 1));
        const int d  = (int)((i >> 11) & (DHEAD - 1));
        const int h  = (int)((i >> 17) & (NHEAD - 1));
        const int b  = (int)(i >> 21);
        vt[i] = qkv[(size_t)(b * SEQ + s_) * QKV_N + 2 * D_MODEL + h * DHEAD + d];
    }
}

// ---------------------------------------------------------------------------
// Tiled WMMA GEMM:  C[M,N] = A[M,K](bf16) @ W[N,K](bf16)^T
// 256 threads = 8 waves, block tile 128x128, BK = 32, wave tile 64x32.
// Double-buffered staging: the next B (weight) tile is DMA'd by the Tensor
// Data Mover (one TENSOR_LOAD_TO_LDS from wave 0, TENSORcnt), and the next A
// (activation) tile by per-lane GLOBAL_LOAD_ASYNC_TO_LDS_B128 (ASYNCcnt),
// both overlapped with the 8 WMMAs/wave on the current tile.
// MODE 0: outBF = bf16(acc)
// MODE 1: outF  = resid + acc                     (residual add, fp32 out)
// MODE 2: outBF = bf16( bf2f(aux) * silu(acc) )   (SwiGLU combine)
// ---------------------------------------------------------------------------
template <int MODE>
__global__ __launch_bounds__(256) void gemm_bf16_kernel(
    const unsigned short* __restrict__ A, const unsigned short* __restrict__ W,
    int M, int N, int K,
    unsigned short* __restrict__ outBF, float* __restrict__ outF,
    const float* __restrict__ resid, const unsigned short* __restrict__ aux) {
    __shared__ __align__(16) unsigned short lA[2][128 * 32];
    __shared__ __align__(16) unsigned short lB[2][128 * 32];

    const int tid  = threadIdx.x;
    const int lane = tid & 31;
    const int wave = tid >> 5;
    const int wm   = (wave & 1) * 64;    // wave M offset in tile
    const int wn   = (wave >> 1) * 32;   // wave N offset in tile
    const int m0   = blockIdx.y * 128;
    const int n0   = blockIdx.x * 128;

    const int ldr = tid >> 1;            // 0..127 : staged row
    const int ldk = (tid & 1) * 16;      // 0 / 16 : k-half

    const v8f Z = {0.f, 0.f, 0.f, 0.f, 0.f, 0.f, 0.f, 0.f};
    v8f acc[4][2];
#pragma unroll
    for (int i = 0; i < 4; ++i)
#pragma unroll
        for (int j = 0; j < 2; ++j) acc[i][j] = Z;

    const unsigned short* aRow = A + (size_t)(m0 + ldr) * K + ldk;
    const unsigned short* bRow = W + (size_t)(n0 + ldr) * K + ldk;

    auto stageA = [&](int buf, int k0) {      // activations: per-lane async DMA
        async_load_b128(&lA[buf][ldr * 32 + ldk],     aRow + k0);
        async_load_b128(&lA[buf][ldr * 32 + ldk + 8], aRow + k0 + 8);
    };
    auto stageB = [&](int buf, int k0) {      // weights: TDM tile DMA
#if HAVE_TDM
        if (wave == 0)
            tdm_load_tile_2d(&lB[buf][0], W + (size_t)n0 * K + k0,
                             /*tile_k=*/32, /*tile_rows=*/128,
                             /*tensor_k=*/(unsigned)K, /*tensor_rows=*/(unsigned)(N - n0));
#else
        async_load_b128(&lB[buf][ldr * 32 + ldk],     bRow + k0);
        async_load_b128(&lB[buf][ldr * 32 + ldk + 8], bRow + k0 + 8);
#endif
    };

    // prologue: DMA first k-tile, wait, publish
    stageA(0, 0);
    stageB(0, 0);
    wait_async0();
    wait_tensor0();
    __syncthreads();

    int buf = 0;
    for (int k0 = 0; k0 < K; k0 += 32) {
        const bool more = (k0 + 32 < K);
        if (more) {                          // overlap DMA with compute
            stageA(buf ^ 1, k0 + 32);
            stageB(buf ^ 1, k0 + 32);
        }

        v16bf af[4], bf[2];
#pragma unroll
        for (int i = 0; i < 4; ++i)
            af[i] = load_a_frag(&lA[buf][(wm + i * 16) * 32], 32, lane);
#pragma unroll
        for (int j = 0; j < 2; ++j)
            bf[j] = load_b_frag(&lB[buf][(wn + j * 16) * 32], 32, lane);
#pragma unroll
        for (int i = 0; i < 4; ++i)
#pragma unroll
            for (int j = 0; j < 2; ++j) acc[i][j] = wmma_bf16(af[i], bf[j], acc[i][j]);

        if (more) {
            wait_async0();                   // this wave's async DMAs landed
            wait_tensor0();                  // TDM done (trivial for waves 1-7)
        }
        __syncthreads();                     // publish buffers workgroup-wide
        buf ^= 1;
    }

    // Epilogue: C layout -> row = r + (lane>=16 ? 8 : 0), col = lane&15
    const int col   = lane & 15;
    const int rowhi = ((lane >> 4) & 1) * 8;
#pragma unroll
    for (int i = 0; i < 4; ++i)
#pragma unroll
        for (int j = 0; j < 2; ++j)
#pragma unroll
            for (int r = 0; r < 8; ++r) {
                const int gm = m0 + wm + i * 16 + r + rowhi;
                const int gn = n0 + wn + j * 16 + col;
                const size_t idx = (size_t)gm * N + gn;
                const float v = acc[i][j][r];
                if (MODE == 0) {
                    outBF[idx] = f2bf(v);
                } else if (MODE == 1) {
                    outF[idx] = resid[idx] + v;
                } else {
                    const float g = v / (1.0f + __expf(-v));   // silu(gate)
                    outBF[idx] = f2bf(bf2f(aux[idx]) * g);
                }
            }
}

// ---------------------------------------------------------------------------
// Flash attention (causal).  Block = 128 thr = 4 independent waves; block
// covers 64 query rows of one (b,h); each wave owns 16 rows.  QK^T and P@V
// both run on v_wmma_f32_16x16x32_bf16; online softmax with shfl_xor row
// reductions in the C-matrix layout; P is re-swizzled to the A layout via a
// 1KB/wave LDS staging tile (in-order DS ops within a wave, no barriers).
// ---------------------------------------------------------------------------
__global__ __launch_bounds__(128) void attn_kernel(const unsigned short* __restrict__ qkv,
                                                   const unsigned short* __restrict__ vt,
                                                   unsigned short* __restrict__ out) {
    __shared__ __align__(16) unsigned short pstage[4][16 * 32];

    const int lane = threadIdx.x & 31;
    const int wave = threadIdx.x >> 5;
    const int b    = blockIdx.z;
    const int h    = blockIdx.y;
    const int qrow = blockIdx.x * 64 + wave * 16;   // this wave's first q row

    // Persistent Q fragments (16 rows x 64 dh  ->  two 16x32 A fragments)
    const unsigned short* qbase = qkv + (size_t)(b * SEQ + qrow) * QKV_N + h * DHEAD;
    const v16bf qa0 = load_a_frag(qbase, QKV_N, lane);
    const v16bf qa1 = load_a_frag(qbase + 32, QKV_N, lane);

    const unsigned short* kbase = qkv + (size_t)(b * SEQ) * QKV_N + D_MODEL + h * DHEAD;
    const unsigned short* vbase = vt + (size_t)(b * NHEAD + h) * DHEAD * SEQ;

    const v8f Z = {0.f, 0.f, 0.f, 0.f, 0.f, 0.f, 0.f, 0.f};
    v8f o[4];
    float mrow[8], lrow[8];
#pragma unroll
    for (int dt = 0; dt < 4; ++dt) o[dt] = Z;
#pragma unroll
    for (int r = 0; r < 8; ++r) { mrow[r] = -1e30f; lrow[r] = 0.f; }

    const int col = lane & 15;
    const int rhi = ((lane >> 4) & 1) * 8;
    unsigned short* ps = pstage[wave];

    for (int kc = 0; kc <= qrow + 15; kc += 32) {
        // ---- scores: two 16x16 tiles (kv kc..+15 and kc+16..+31) ----
        v8f s0 = Z, s1 = Z;
        {
            const unsigned short* k0p = kbase + (size_t)kc * QKV_N;
            const unsigned short* k1p = kbase + (size_t)(kc + 16) * QKV_N;
            s0 = wmma_bf16(qa0, load_b_frag(k0p, QKV_N, lane), s0);
            s0 = wmma_bf16(qa1, load_b_frag(k0p + 32, QKV_N, lane), s0);
            s1 = wmma_bf16(qa0, load_b_frag(k1p, QKV_N, lane), s1);
            s1 = wmma_bf16(qa1, load_b_frag(k1p + 32, QKV_N, lane), s1);
        }
        // ---- scale + causal mask + online softmax update ----
#pragma unroll
        for (int r = 0; r < 8; ++r) {
            const int qg = qrow + r + rhi;
            float x0 = (kc + col      <= qg) ? s0[r] * 0.125f : -1e30f;
            float x1 = (kc + 16 + col <= qg) ? s1[r] * 0.125f : -1e30f;
            float mx = fmaxf(x0, x1);
#pragma unroll
            for (int msk = 1; msk < 16; msk <<= 1) mx = fmaxf(mx, __shfl_xor(mx, msk, 32));
            const float mn   = fmaxf(mrow[r], mx);
            const float corr = __expf(mrow[r] - mn);
            const float p0   = __expf(x0 - mn);
            const float p1   = __expf(x1 - mn);
            float sum = p0 + p1;
#pragma unroll
            for (int msk = 1; msk < 16; msk <<= 1) sum += __shfl_xor(sum, msk, 32);
            lrow[r] = lrow[r] * corr + sum;
            mrow[r] = mn;
#pragma unroll
            for (int dt = 0; dt < 4; ++dt) o[dt][r] *= corr;
            const int prow = r + rhi;
            ps[prow * 32 + col]      = f2bf(p0);
            ps[prow * 32 + 16 + col] = f2bf(p1);
        }
        // ---- P (16x32) @ V (32x64): A frag from LDS, B frags from vt ----
        const v16bf pa = load_a_frag(ps, 32, lane);
#pragma unroll
        for (int dt = 0; dt < 4; ++dt) {
            const v16bf vb = load_b_frag(vbase + (size_t)dt * 16 * SEQ + kc, SEQ, lane);
            o[dt] = wmma_bf16(pa, vb, o[dt]);
        }
    }

    // ---- epilogue: O /= l, write bf16 ----
    unsigned short* ob = out + (size_t)(b * SEQ + qrow) * D_MODEL + h * DHEAD;
#pragma unroll
    for (int dt = 0; dt < 4; ++dt)
#pragma unroll
        for (int r = 0; r < 8; ++r)
            ob[(size_t)(r + rhi) * D_MODEL + dt * 16 + col] = f2bf(o[dt][r] / lrow[r]);
}

// ---------------------------------------------------------------------------
// Launcher
// ---------------------------------------------------------------------------
extern "C" void kernel_launch(void* const* d_in, const int* in_sizes, int n_in,
                              void* d_out, int out_size, void* d_ws, size_t ws_size,
                              hipStream_t stream) {
    (void)in_sizes; (void)n_in; (void)out_size; (void)ws_size;
    const float* x      = (const float*)d_in[0];
    const float* w_qkv  = (const float*)d_in[1];
    const float* w_o    = (const float*)d_in[2];
    const float* w_up   = (const float*)d_in[3];
    const float* w_gate = (const float*)d_in[4];
    const float* w_down = (const float*)d_in[5];
    const float* scale1 = (const float*)d_in[6];
    const float* scale2 = (const float*)d_in[7];
    float* out = (float*)d_out;

    // Workspace layout (256B aligned, bf16 unless noted)
    char* ws = (char*)d_ws;
    size_t off = 0;
    auto take = [&](size_t bytes) {
        char* p = ws + off;
        off += (bytes + 255) & ~(size_t)255;
        return p;
    };
    unsigned short* wq   = (unsigned short*)take((size_t)QKV_N * D_MODEL * 2);   //  6 MB
    unsigned short* wo   = (unsigned short*)take((size_t)D_MODEL * D_MODEL * 2); //  2 MB
    unsigned short* wup  = (unsigned short*)take((size_t)HIDDEN * D_MODEL * 2);  //  8 MB
    unsigned short* wg   = (unsigned short*)take((size_t)HIDDEN * D_MODEL * 2);  //  8 MB
    unsigned short* wd   = (unsigned short*)take((size_t)D_MODEL * HIDDEN * 2);  //  8 MB
    unsigned short* xn   = (unsigned short*)take((size_t)MTOK * D_MODEL * 2);    //  8 MB (xn1/xn2)
    unsigned short* qkvb = (unsigned short*)take((size_t)MTOK * QKV_N * 2);      // 24 MB
    unsigned short* vtb  = (unsigned short*)take((size_t)BATCH * NHEAD * DHEAD * SEQ * 2); // 8 MB
    unsigned short* attb = (unsigned short*)take((size_t)MTOK * D_MODEL * 2);    //  8 MB
    float*          h1   = (float*)take((size_t)MTOK * D_MODEL * 4);             // 16 MB
    unsigned short* actb = (unsigned short*)take((size_t)MTOK * HIDDEN * 2);     // 32 MB
    // up-projection reuses the (dead) qkv+vt region: 24 MB + 8 MB = 32 MB exactly
    unsigned short* upb  = qkvb;

    // 1) weights fp32 -> bf16
    cvt_f32_bf16_kernel<<<2048, 256, 0, stream>>>(w_qkv,  wq,  QKV_N * D_MODEL);
    cvt_f32_bf16_kernel<<<2048, 256, 0, stream>>>(w_o,    wo,  D_MODEL * D_MODEL);
    cvt_f32_bf16_kernel<<<2048, 256, 0, stream>>>(w_up,   wup, HIDDEN * D_MODEL);
    cvt_f32_bf16_kernel<<<2048, 256, 0, stream>>>(w_gate, wg,  HIDDEN * D_MODEL);
    cvt_f32_bf16_kernel<<<2048, 256, 0, stream>>>(w_down, wd,  D_MODEL * HIDDEN);

    // 2) xn1 = rmsnorm(x) (bf16)
    rmsnorm_kernel<<<MTOK, 256, 0, stream>>>(x, scale1, xn);

    // 3) qkv = xn1 @ w_qkv^T  (bf16 out)
    gemm_bf16_kernel<0><<<dim3(QKV_N / 128, MTOK / 128), 256, 0, stream>>>(
        xn, wq, MTOK, QKV_N, D_MODEL, qkvb, nullptr, nullptr, nullptr);

    // 4) vt[b,h,d,s] = V
    transpose_v_kernel<<<2048, 256, 0, stream>>>(qkvb, vtb);

    // 5) flash attention -> attb (bf16)
    attn_kernel<<<dim3(SEQ / 64, NHEAD, BATCH), 128, 0, stream>>>(qkvb, vtb, attb);

    // 6) h1 = x + attb @ w_o^T  (fp32)
    gemm_bf16_kernel<1><<<dim3(D_MODEL / 128, MTOK / 128), 256, 0, stream>>>(
        attb, wo, MTOK, D_MODEL, D_MODEL, nullptr, h1, x, nullptr);

    // 7) xn2 = rmsnorm(h1)
    rmsnorm_kernel<<<MTOK, 256, 0, stream>>>(h1, scale2, xn);

    // 8) up = xn2 @ w_up^T (bf16)
    gemm_bf16_kernel<0><<<dim3(HIDDEN / 128, MTOK / 128), 256, 0, stream>>>(
        xn, wup, MTOK, HIDDEN, D_MODEL, upb, nullptr, nullptr, nullptr);

    // 9) act = up * silu(xn2 @ w_gate^T) (bf16)
    gemm_bf16_kernel<2><<<dim3(HIDDEN / 128, MTOK / 128), 256, 0, stream>>>(
        xn, wg, MTOK, HIDDEN, D_MODEL, actb, nullptr, nullptr, upb);

    // 10) out = h1 + act @ w_down^T (fp32)
    gemm_bf16_kernel<1><<<dim3(D_MODEL / 128, MTOK / 128), 256, 0, stream>>>(
        actb, wd, MTOK, D_MODEL, HIDDEN, nullptr, out, h1, nullptr);
}
